// WindowAttentionNoOverlap_35132832481954
// MI455X (gfx1250) — matrix operand order repaired and checked by reading, not compiled, not run on hardware
//
#include <hip/hip_runtime.h>
#include <hip/hip_bf16.h>

typedef _Float16 f16;
typedef __attribute__((ext_vector_type(8)))  _Float16 v8h;
typedef __attribute__((ext_vector_type(16))) _Float16 v16h;
typedef __attribute__((ext_vector_type(8)))  float    v8f;

#define B_  2
#define S_  8192
#define H_  8
#define D_  64
#define W_  256
#define C_  32            // S_/W_ chunks
#define KW_ 768           // 3*W_ window
#define NW_ 4             // waves per block
#define BLK_ 128          // threads per block

// LDS partition (bytes)
#define Q16_OFF 0                         // 64*64 f16      =   8192
#define KV_OFF  8192                      // 768*64 f16     =  98304 (K row-major, then V d-major)
#define SC_OFF  (8192 + 98304)            // 4*16*768 f32   = 196608
#define SMEM_BYTES (8192 + 98304 + 196608) // 303104 <= 320KB

__device__ __forceinline__ v16h cat8(v8h lo, v8h hi) {
    return __builtin_shufflevector(lo, hi, 0,1,2,3,4,5,6,7,8,9,10,11,12,13,14,15);
}

__global__ void WindowAttentionNoOverlap_35132832481954_kernel(
    const float* __restrict__ qg, const float* __restrict__ kg,
    const float* __restrict__ vg, float* __restrict__ out,
    float* __restrict__ att)
{
    extern __shared__ char smem[];
    f16*   q16    = (f16*)(smem + Q16_OFF);   // [64][64]
    f16*   kv16   = (f16*)(smem + KV_OFF);    // K: [768][64]; later V^T: [64][768]
    float* scores = (float*)(smem + SC_OFF);  // [4*16][768]

    int g = blockIdx.x;
    const int slice = g & 3;  g >>= 2;        // quarter-chunk (64 query rows)
    const int cc    = g & 31; g >>= 5;        // chunk
    const int hh    = g & 7;  g >>= 3;        // head
    const int bb    = g;                      // batch

    const int tid  = threadIdx.x;
    const int wave = tid >> 5;
    const int lane = tid & 31;
    const int qrow_base = cc * W_ + slice * 64;

    // ---------- Phase 1: load Q (64x64) and K (768x64, zero-padded) as f16 ----------
    #pragma unroll
    for (int i = 0; i < 8; ++i) {             // 64*16 float4 / 128 threads
        int e4  = tid + i * BLK_;
        int row = e4 >> 4;
        int c4  = (e4 & 15) << 2;
        const float4 x = *(const float4*)(qg + ((size_t)((bb * S_ + qrow_base + row) * H_ + hh)) * D_ + c4);
        f16* d = q16 + row * 64 + c4;
        d[0] = (f16)x.x; d[1] = (f16)x.y; d[2] = (f16)x.z; d[3] = (f16)x.w;
    }
    for (int i = 0; i < 96; ++i) {            // 768*16 float4 / 128 threads
        int e4 = tid + i * BLK_;
        int kr = e4 >> 4;                     // 0..767 window row
        int c4 = (e4 & 15) << 2;
        int ch = cc + (kr >> 8) - 1;          // neighbor chunk index
        float4 x = make_float4(0.f, 0.f, 0.f, 0.f);
        if (ch >= 0 && ch < C_) {
            int srow = ch * W_ + (kr & 255);
            x = *(const float4*)(kg + ((size_t)((bb * S_ + srow) * H_ + hh)) * D_ + c4);
        }
        f16* d = kv16 + kr * 64 + c4;
        d[0] = (f16)x.x; d[1] = (f16)x.y; d[2] = (f16)x.z; d[3] = (f16)x.w;
    }
    // Warm V into L2 while we compute QK (global_prefetch_b8)
    #pragma unroll
    for (int i = 0; i < 6; ++i) {
        int kr = tid + i * BLK_;              // one row (256B) per prefetch
        int ch = cc + (kr >> 8) - 1;
        if (ch >= 0 && ch < C_) {
            int srow = ch * W_ + (kr & 255);
            __builtin_prefetch(vg + ((size_t)((bb * S_ + srow) * H_ + hh)) * D_, 0, 1);
        }
    }
    __syncthreads();

    // ---------- Phase 2: GEMM1  S(16x768) = Q(16x64) * K^T per wave ----------
    // Two column tiles per iteration with independent accumulators: back-to-back
    // WMMAs are independent, so the f16 WMMA->WMMA hazard slots hold useful work
    // instead of compiler-inserted v_nops.
    const int laneq = lane & 15;
    const int koff  = (lane < 16) ? 0 : 16;   // A-matrix k sub-offset (contiguous k = 16h + s)
    const int boff  = (lane < 16) ? 0 : 16;   // B-matrix k sub-offset (contiguous k = 16h + s)

    const f16* qrow = q16 + (wave * 16 + laneq) * 64;
    const v16h a0 = cat8(*(const v8h*)(qrow + koff),      *(const v8h*)(qrow + koff + 8));
    const v16h a1 = cat8(*(const v8h*)(qrow + 32 + koff), *(const v8h*)(qrow + 32 + koff + 8));

    float* sw = scores + wave * 16 * KW_;
    const int r0 = (lane < 16) ? 0 : 8;
    for (int n = 0; n < 48; n += 2) {
        const f16* kbA = kv16 + ((n + 0) * 16 + laneq) * 64 + boff;
        const f16* kbB = kv16 + ((n + 1) * 16 + laneq) * 64 + boff;
        v16h b0A = cat8(*(const v8h*)(kbA),      *(const v8h*)(kbA + 8));
        v16h b1A = cat8(*(const v8h*)(kbA + 32), *(const v8h*)(kbA + 40));
        v16h b0B = cat8(*(const v8h*)(kbB),      *(const v8h*)(kbB + 8));
        v16h b1B = cat8(*(const v8h*)(kbB + 32), *(const v8h*)(kbB + 40));
        v8f accA = {}, accB = {};
        accA = __builtin_amdgcn_wmma_f32_16x16x32_f16(false, a0, false, b0A, (short)0, accA, false, false);
        accB = __builtin_amdgcn_wmma_f32_16x16x32_f16(false, a0, false, b0B, (short)0, accB, false, false);
        accA = __builtin_amdgcn_wmma_f32_16x16x32_f16(false, a1, false, b1A, (short)0, accA, false, false);
        accB = __builtin_amdgcn_wmma_f32_16x16x32_f16(false, a1, false, b1B, (short)0, accB, false, false);
        const int colA = (n + 0) * 16 + laneq;
        const int colB = (n + 1) * 16 + laneq;
        #pragma unroll
        for (int r = 0; r < 8; ++r) sw[(r0 + r) * KW_ + colA] = accA[r];
        #pragma unroll
        for (int r = 0; r < 8; ++r) sw[(r0 + r) * KW_ + colB] = accB[r];
    }

    // ---------- Phase 3: softmax over 768 (includes zero-padded slots, like ref) ----------
    {
        const int row  = lane >> 1;
        const int half = lane & 1;
        float4* p = (float4*)(scores + (wave * 16 + row) * KW_ + half * 384);
        float m = -3.4e38f;
        for (int j = 0; j < 96; ++j) {
            float4 x = p[j];
            m = fmaxf(m, fmaxf(fmaxf(x.x, x.y), fmaxf(x.z, x.w)));
        }
        m = fmaxf(m, __shfl_xor(m, 1));
        float ssum = 0.f;
        for (int j = 0; j < 96; ++j) {
            float4 x = p[j];
            x.x = __expf(x.x - m); x.y = __expf(x.y - m);
            x.z = __expf(x.z - m); x.w = __expf(x.w - m);
            p[j] = x;
            ssum += (x.x + x.y) + (x.z + x.w);
        }
        ssum += __shfl_xor(ssum, 1);
        const float inv = 1.0f / ssum;
        for (int j = 0; j < 96; ++j) {
            float4 x = p[j];
            x.x *= inv; x.y *= inv; x.z *= inv; x.w *= inv;
            p[j] = x;
        }
    }
    // coalesced write-out of probabilities (dominant HBM traffic)
    for (int r = 0; r < 16; ++r) {
        const float4* src = (const float4*)(scores + (wave * 16 + r) * KW_);
        const int grow = qrow_base + wave * 16 + r;
        float4* dst = (float4*)(att + ((size_t)((bb * S_ + grow) * H_ + hh)) * KW_);
        for (int j = lane; j < 192; j += 32) dst[j] = src[j];
    }
    __syncthreads();   // all waves done reading K before V overwrites kv16

    // ---------- Phase 4: load V transposed (d-major) into kv16 ----------
    for (int i = 0; i < 96; ++i) {
        int e4 = tid + i * BLK_;
        int kr = e4 >> 4;
        int c4 = (e4 & 15) << 2;
        int ch = cc + (kr >> 8) - 1;
        float4 x = make_float4(0.f, 0.f, 0.f, 0.f);
        if (ch >= 0 && ch < C_) {
            int srow = ch * W_ + (kr & 255);
            x = *(const float4*)(vg + ((size_t)((bb * S_ + srow) * H_ + hh)) * D_ + c4);
        }
        kv16[(c4 + 0) * KW_ + kr] = (f16)x.x;
        kv16[(c4 + 1) * KW_ + kr] = (f16)x.y;
        kv16[(c4 + 2) * KW_ + kr] = (f16)x.z;
        kv16[(c4 + 3) * KW_ + kr] = (f16)x.w;
    }
    __syncthreads();

    // ---------- Phase 5: GEMM2  O(16x64) = P(16x768) * V(768x64) ----------
    // 4 independent accumulators round-robin: no WMMA->WMMA hazard stalls.
    const float* prow = scores + (wave * 16 + laneq) * KW_;
    v8f o[4] = {};
    for (int ks = 0; ks < 24; ++ks) {
        const int k0 = ks * 32;
        const float* pa = prow + k0 + koff;
        v16h a;
        #pragma unroll
        for (int e = 0; e < 8; ++e) a[e]     = (f16)pa[e];
        #pragma unroll
        for (int e = 0; e < 8; ++e) a[8 + e] = (f16)pa[8 + e];
        #pragma unroll
        for (int nt = 0; nt < 4; ++nt) {
            const f16* vb = kv16 + (nt * 16 + laneq) * KW_ + k0 + boff;
            v16h b = cat8(*(const v8h*)(vb), *(const v8h*)(vb + 8));
            o[nt] = __builtin_amdgcn_wmma_f32_16x16x32_f16(false, a, false, b, (short)0, o[nt], false, false);
        }
    }
    #pragma unroll
    for (int nt = 0; nt < 4; ++nt) {
        const int col = nt * 16 + laneq;
        #pragma unroll
        for (int r = 0; r < 8; ++r) {
            const int grow = qrow_base + wave * 16 + r0 + r;
            out[((size_t)((bb * S_ + grow) * H_ + hh)) * D_ + col] = o[nt][r];
        }
    }
}

extern "C" void kernel_launch(void* const* d_in, const int* in_sizes, int n_in,
                              void* d_out, int out_size, void* d_ws, size_t ws_size,
                              hipStream_t stream) {
    const float* q = (const float*)d_in[0];
    const float* k = (const float*)d_in[1];
    const float* v = (const float*)d_in[2];
    float* out = (float*)d_out;                                   // (b,s,h,d)
    float* att = (float*)d_out + (size_t)B_ * S_ * H_ * D_;       // (b,s,h,3W)

    (void)in_sizes; (void)n_in; (void)out_size; (void)d_ws; (void)ws_size;

    (void)hipFuncSetAttribute((const void*)WindowAttentionNoOverlap_35132832481954_kernel,
                              hipFuncAttributeMaxDynamicSharedMemorySize, SMEM_BYTES);

    const int blocks = B_ * H_ * C_ * 4;   // 2048
    WindowAttentionNoOverlap_35132832481954_kernel<<<blocks, BLK_, SMEM_BYTES, stream>>>(
        q, k, v, out, att);
}
